// Attention_63024350101513
// MI455X (gfx1250) — matrix operand order repaired and checked by reading, not compiled, not run on hardware
//
#include <hip/hip_runtime.h>
#include <hip/hip_bf16.h>

typedef __attribute__((ext_vector_type(16))) __bf16 bf16x16;
typedef __attribute__((ext_vector_type(8)))  __bf16 bf16x8;
typedef __attribute__((ext_vector_type(8)))  float  f32x8;

constexpr int kBATCH = 8;
constexpr int kSEQ   = 2048;
constexpr int kDIM   = 512;
constexpr int kHEADS = 8;
constexpr int kDH    = 64;
constexpr int kM     = kBATCH * kSEQ;   // 16384 rows

// ---------------------------------------------------------------------------
// WMMA helpers (CDNA5 bf16, 16x16x32, fp32 accumulate)
// ---------------------------------------------------------------------------
__device__ __forceinline__ f32x8 wmma_bf16(bf16x16 a, bf16x16 b, f32x8 c) {
    return __builtin_amdgcn_wmma_f32_16x16x32_bf16(
        /*neg_a=*/false, a, /*neg_b=*/false, b,
        /*c_mod=*/(short)0, c, /*reuse_a=*/false, /*reuse_b=*/false);
}

// A-matrix fragment (16x32 bf16). Lane l holds row l&15;
// element i -> K = (i<8) ? ((l>>4)*8 + i) : (16 + (l>>4)*8 + i-8)
__device__ __forceinline__ bf16x16 frag_ld_a(const __bf16* base, int ld, int row0, int k0) {
    int lane = threadIdx.x & 31;
    const __bf16* p = base + (row0 + (lane & 15)) * ld + k0 + ((lane >> 4) << 3);
    bf16x8 lo = *(const bf16x8*)(p);        // K = kb .. kb+7
    bf16x8 hi = *(const bf16x8*)(p + 16);   // K = kb+16 .. kb+23
    return __builtin_shufflevector(lo, hi, 0,1,2,3,4,5,6,7,8,9,10,11,12,13,14,15);
}

// B-matrix fragment (32x16 bf16). Lane l holds col l&15;
// element i -> K = (i<8) ? ((l>>4)*8 + i) : (16 + (l>>4)*8 + i-8)
__device__ __forceinline__ bf16x16 frag_ld_b(const __bf16* base, int ld, int col0, int k0) {
    int lane = threadIdx.x & 31;
    const __bf16* p = base + (col0 + (lane & 15)) * ld + k0 + ((lane >> 4) << 3);
    bf16x8 lo = *(const bf16x8*)(p);
    bf16x8 hi = *(const bf16x8*)(p + 16);
    return __builtin_shufflevector(lo, hi, 0,1,2,3,4,5,6,7,8,9,10,11,12,13,14,15);
}

// ---------------------------------------------------------------------------
// Precision-conversion kernels
// ---------------------------------------------------------------------------
__global__ void cvt_f32_bf16(const float* __restrict__ in, __bf16* __restrict__ out, int n) {
    int i = blockIdx.x * blockDim.x + threadIdx.x;
    if (i < n) out[i] = (__bf16)in[i];
}

// Weight conversion: fold sel into output channel, store TRANSPOSED [n][k]
// so the GEMM B-tile is K-contiguous (no in-loop transpose scatter).
__global__ void cvt_w_t_bf16(const float* __restrict__ W, const float* __restrict__ sel,
                             __bf16* __restrict__ outT) {
    int i = blockIdx.x * blockDim.x + threadIdx.x;   // over 512*512, coalesced read
    int k = i >> 9, n = i & (kDIM - 1);
    float s = sel ? sel[n] : 1.0f;
    outT[(size_t)n * kDIM + k] = (__bf16)(W[i] * s);
}

// ---------------------------------------------------------------------------
// GEMM: C[M,512] = A[M,512](bf16) * Wt[512n][512k](bf16, K-contiguous)
// mode 0: bf16 out remapped to [B,H,N,64]   (Q, K projections)
// mode 1: fp32 out [M,512] + bias           (output projection)
// mode 2: bf16 out transposed [B*H, 64, N]  (V projection, for flash V^T)
// Block: 256 threads (8 waves). Tile 128x128, K-chunk 64, reg double-buffer.
// ---------------------------------------------------------------------------
__global__ __launch_bounds__(256) void gemm512(
    const __bf16* __restrict__ A, const __bf16* __restrict__ Bt,
    const float* __restrict__ bias, void* __restrict__ Out, int mode)
{
    __shared__ __align__(16) __bf16 As[128][72];   // [m][k], +8 pad
    __shared__ __align__(16) __bf16 Bs[128][72];   // [n][k], +8 pad

    const int tid  = threadIdx.x;
    const int lane = tid & 31;
    const int w    = tid >> 5;       // 0..7
    const int wm   = w >> 2;         // 0..1  (64-row slab)
    const int wn   = w & 3;          // 0..3  (32-col slab)
    const int m0   = blockIdx.y * 128;
    const int n0   = blockIdx.x * 128;

    f32x8 acc[4][2];
    f32x8 zero = {};
    #pragma unroll
    for (int i = 0; i < 4; i++)
        #pragma unroll
        for (int j = 0; j < 2; j++) acc[i][j] = zero;

    // tile-copy mapping: 2 threads per row, each owns 32 contiguous K
    const int crow = tid >> 1;            // 0..127
    const int cseg = (tid & 1) * 32;      // 0 / 32

    const __bf16* gA = A  + (size_t)(m0 + crow) * kDIM + cseg;
    const __bf16* gB = Bt + (size_t)(n0 + crow) * kDIM + cseg;

    // preload chunk 0 into registers
    bf16x8 ra[4], rb[4];
    #pragma unroll
    for (int j = 0; j < 4; j++) {
        ra[j] = *(const bf16x8*)(gA + j * 8);
        rb[j] = *(const bf16x8*)(gB + j * 8);
    }

    for (int kt = 0; kt < kDIM; kt += 64) {
        __syncthreads();   // everyone done reading previous tiles
        #pragma unroll
        for (int j = 0; j < 4; j++) {
            *(bf16x8*)&As[crow][cseg + j * 8] = ra[j];
            *(bf16x8*)&Bs[crow][cseg + j * 8] = rb[j];
        }
        __syncthreads();
        if (kt + 64 < kDIM) {  // issue next chunk's loads; overlap with WMMAs
            #pragma unroll
            for (int j = 0; j < 4; j++) {
                ra[j] = *(const bf16x8*)(gA + kt + 64 + j * 8);
                rb[j] = *(const bf16x8*)(gB + kt + 64 + j * 8);
            }
            if (kt + 128 < kDIM)
                __builtin_prefetch(gA + kt + 128, 0, 1);   // global_prefetch_b8
        }

        #pragma unroll
        for (int kk = 0; kk < 64; kk += 32) {
            bf16x16 afr[4];
            #pragma unroll
            for (int mi = 0; mi < 4; mi++)
                afr[mi] = frag_ld_a(&As[0][0], 72, wm * 64 + mi * 16, kk);
            #pragma unroll
            for (int ni = 0; ni < 2; ni++) {
                bf16x16 bfr = frag_ld_b(&Bs[0][0], 72, wn * 32 + ni * 16, kk);
                #pragma unroll
                for (int mi = 0; mi < 4; mi++)
                    acc[mi][ni] = wmma_bf16(afr[mi], bfr, acc[mi][ni]);
            }
        }
    }

    // Epilogue. C layout: VGPR r -> row r + 8*(lane>>4), col = lane&15.
    #pragma unroll
    for (int mi = 0; mi < 4; mi++) {
        #pragma unroll
        for (int ni = 0; ni < 2; ni++) {
            const int col     = n0 + wn * 32 + ni * 16 + (lane & 15);
            const int rowbase = m0 + wm * 64 + mi * 16 + ((lane >> 4) << 3);
            if (mode == 0) {
                __bf16* outb = (__bf16*)Out;
                const int h = col >> 6, d = col & 63;
                #pragma unroll
                for (int r = 0; r < 8; r++) {
                    const int m = rowbase + r;
                    const int b = m >> 11, n = m & (kSEQ - 1);
                    outb[(((size_t)(b * kHEADS + h) * kSEQ + n) * kDH) + d] =
                        (__bf16)acc[mi][ni][r];
                }
            } else if (mode == 1) {
                float* outf = (float*)Out;
                const float bv = bias[col];
                #pragma unroll
                for (int r = 0; r < 8; r++)
                    outf[(size_t)(rowbase + r) * kDIM + col] = acc[mi][ni][r] + bv;
            } else {  // mode 2: V^T layout [b*H+h][d][n]; 8 rows = 8 consecutive n
                __bf16* outb = (__bf16*)Out;
                const int h = col >> 6, d = col & 63;
                const int b = rowbase >> 11, nb = rowbase & (kSEQ - 1);
                bf16x8 v;
                #pragma unroll
                for (int r = 0; r < 8; r++) v[r] = (__bf16)acc[mi][ni][r];
                *(bf16x8*)&outb[((size_t)(b * kHEADS + h) * kDH + d) * kSEQ + nb] = v;
            }
        }
    }
}

// ---------------------------------------------------------------------------
// Flash attention: one block per (64-row Q slab, b*h). 128 threads = 4 waves,
// each wave owns 16 Q rows. Streams 64-key blocks with register double-buffer.
// Q/K: bf16 [B*H][N][64]; Vt: bf16 [B*H][64][N]; O: bf16 [B][N][512].
// ---------------------------------------------------------------------------
__global__ __launch_bounds__(128) void flash64(
    const __bf16* __restrict__ Q, const __bf16* __restrict__ K,
    const __bf16* __restrict__ Vt, __bf16* __restrict__ O)
{
    __shared__ __align__(16) __bf16 Qs[64][72];      // [row][d], pre-scaled
    __shared__ __align__(16) __bf16 Ks[64][72];      // [key][d]
    __shared__ __align__(16) __bf16 Vts[64][72];     // [d][key]
    __shared__ __align__(16) __bf16 Ps[4][16][72];   // per-wave P, [row][key]
    __shared__ __align__(16) float  Ss[4][16][72];   // per-wave fp32 scores
    __shared__ float alphas[4][16];
    __shared__ float lrow[4][16];

    const int tid  = threadIdx.x;
    const int lane = tid & 31;
    const int w    = tid >> 5;             // wave 0..3
    const int q0   = blockIdx.x * 64;
    const int bh   = blockIdx.y;
    const __bf16* Qp = Q  + (size_t)bh * kSEQ * kDH;
    const __bf16* Kp = K  + (size_t)bh * kSEQ * kDH;
    const __bf16* Vp = Vt + (size_t)bh * kDH * kSEQ;

    const int crow = tid >> 1;         // 0..63
    const int cseg = (tid & 1) * 32;   // 0 / 32

    {   // load Q slab, fold softmax scale (1/sqrt(64) = 0.125, exact in bf16)
        const __bf16* gp = Qp + (size_t)(q0 + crow) * kDH + cseg;
        #pragma unroll
        for (int j = 0; j < 4; j++) {
            bf16x8 v = *(const bf16x8*)(gp + j * 8);
            #pragma unroll
            for (int i = 0; i < 8; i++)
                Qs[crow][cseg + j * 8 + i] = (__bf16)((float)v[i] * 0.125f);
        }
    }

    // K rows are [key][d] (d-contig); V^T rows are [d][key] (key-contig)
    const __bf16* gK = Kp + (size_t)crow * kDH + cseg;   // advance by key block
    const __bf16* gV = Vp + (size_t)crow * kSEQ + cseg;  // advance by key block

    bf16x8 kreg[4], vreg[4];
    #pragma unroll
    for (int j = 0; j < 4; j++) {       // preload key block 0
        kreg[j] = *(const bf16x8*)(gK + j * 8);
        vreg[j] = *(const bf16x8*)(gV + j * 8);
    }

    f32x8 oacc[4];
    f32x8 zero = {};
    #pragma unroll
    for (int i = 0; i < 4; i++) oacc[i] = zero;

    const int sr = lane & 15;     // softmax row within wave slab
    const int sh = lane >> 4;     // column half (0: cols 0-31, 1: cols 32-63)
    float m_run = -3.0e38f, l_run = 0.0f;

    for (int kb = 0; kb < kSEQ; kb += 64) {
        __syncthreads();   // previous iteration finished reading Ks/Vts
        #pragma unroll
        for (int j = 0; j < 4; j++) {
            *(bf16x8*)&Ks[crow][cseg + j * 8]  = kreg[j];
            *(bf16x8*)&Vts[crow][cseg + j * 8] = vreg[j];
        }
        __syncthreads();
        if (kb + 64 < kSEQ) {  // overlap next block's global loads with compute
            #pragma unroll
            for (int j = 0; j < 4; j++) {
                kreg[j] = *(const bf16x8*)(gK + (size_t)(kb + 64) * kDH + j * 8);
                vreg[j] = *(const bf16x8*)(gV + (kb + 64) + j * 8);
            }
        }

        // S = (Q*scale) K^T : 4 col-tiles x (K-dim 64 = 2 WMMA steps)
        bf16x16 qa0 = frag_ld_a(&Qs[0][0], 72, w * 16, 0);
        bf16x16 qa1 = frag_ld_a(&Qs[0][0], 72, w * 16, 32);
        #pragma unroll
        for (int ct = 0; ct < 4; ct++) {
            f32x8 s = {};
            s = wmma_bf16(qa0, frag_ld_b(&Ks[0][0], 72, ct * 16, 0),  s);
            s = wmma_bf16(qa1, frag_ld_b(&Ks[0][0], 72, ct * 16, 32), s);
            #pragma unroll
            for (int r = 0; r < 8; r++)
                Ss[w][r + sh * 8][ct * 16 + (lane & 15)] = s[r];
        }

        // Online softmax: lane handles row sr, cols [sh*32, sh*32+32)
        float mx = -3.0e38f;
        #pragma unroll
        for (int j = 0; j < 32; j++) mx = fmaxf(mx, Ss[w][sr][sh * 32 + j]);
        mx = fmaxf(mx, __shfl_xor(mx, 16, 32));
        const float m_new = fmaxf(m_run, mx);
        const float alpha = __expf(m_run - m_new);
        float ssum = 0.0f;
        #pragma unroll
        for (int j = 0; j < 32; j++) {
            const float p = __expf(Ss[w][sr][sh * 32 + j] - m_new);
            Ps[w][sr][sh * 32 + j] = (__bf16)p;
            ssum += p;
        }
        ssum += __shfl_xor(ssum, 16, 32);
        l_run = l_run * alpha + ssum;
        m_run = m_new;
        if (sh == 0) alphas[w][sr] = alpha;

        // rescale running O by alpha (per accumulator row)
        float al[8];
        #pragma unroll
        for (int r = 0; r < 8; r++) al[r] = alphas[w][r + sh * 8];
        #pragma unroll
        for (int dt = 0; dt < 4; dt++)
            #pragma unroll
            for (int r = 0; r < 8; r++) oacc[dt][r] *= al[r];

        // O += P V
        bf16x16 pa0 = frag_ld_a(&Ps[w][0][0], 72, 0, 0);
        bf16x16 pa1 = frag_ld_a(&Ps[w][0][0], 72, 0, 32);
        #pragma unroll
        for (int dt = 0; dt < 4; dt++) {
            oacc[dt] = wmma_bf16(pa0, frag_ld_b(&Vts[0][0], 72, dt * 16, 0),  oacc[dt]);
            oacc[dt] = wmma_bf16(pa1, frag_ld_b(&Vts[0][0], 72, dt * 16, 32), oacc[dt]);
        }
    }

    // normalize by 1/l and store to [B][N][512] at head offset
    if (sh == 0) lrow[w][sr] = l_run;
    float inv[8];
    #pragma unroll
    for (int r = 0; r < 8; r++) inv[r] = 1.0f / lrow[w][r + sh * 8];

    const int bb = bh >> 3, h = bh & 7;
    const int colbase = h * kDH + (lane & 15);
    #pragma unroll
    for (int dt = 0; dt < 4; dt++) {
        const int col = colbase + dt * 16;
        #pragma unroll
        for (int r = 0; r < 8; r++) {
            const int n = q0 + w * 16 + r + sh * 8;
            O[((size_t)bb * kSEQ + n) * kDIM + col] = (__bf16)(oacc[dt][r] * inv[r]);
        }
    }
}

// ---------------------------------------------------------------------------
// Launcher. Inputs: x, Wq, Wk, Wv, sel, Wo, bo (all fp32). Output fp32.
// ---------------------------------------------------------------------------
extern "C" void kernel_launch(void* const* d_in, const int* in_sizes, int n_in,
                              void* d_out, int out_size, void* d_ws, size_t ws_size,
                              hipStream_t stream) {
    const float* x   = (const float*)d_in[0];
    const float* Wq  = (const float*)d_in[1];
    const float* Wk  = (const float*)d_in[2];
    const float* Wv  = (const float*)d_in[3];
    const float* sel = (const float*)d_in[4];
    const float* Wo  = (const float*)d_in[5];
    const float* bo  = (const float*)d_in[6];
    float* out = (float*)d_out;

    const size_t SZ = (size_t)kM * kDIM * sizeof(__bf16);   // 16 MiB per tensor
    const size_t WS = (size_t)kDIM * kDIM * sizeof(__bf16); // 512 KiB per weight
    char* ws = (char*)d_ws;
    __bf16* xb  = (__bf16*)(ws);
    __bf16* Qb  = (__bf16*)(ws + 1 * SZ);
    __bf16* Kb  = (__bf16*)(ws + 2 * SZ);
    __bf16* Vb  = (__bf16*)(ws + 3 * SZ);      // transposed [B*H][64][N]
    __bf16* Ob  = (__bf16*)(ws + 4 * SZ);
    __bf16* Wqb = (__bf16*)(ws + 5 * SZ);
    __bf16* Wkb = (__bf16*)(ws + 5 * SZ + 1 * WS);
    __bf16* Wvb = (__bf16*)(ws + 5 * SZ + 2 * WS);
    __bf16* Wob = (__bf16*)(ws + 5 * SZ + 3 * WS);

    const int nx = kM * kDIM;                 // 8,388,608
    const int nw = kDIM * kDIM;               // 262,144
    cvt_f32_bf16<<<dim3(nx / 256), 256, 0, stream>>>(x, xb, nx);
    cvt_w_t_bf16<<<dim3(nw / 256), 256, 0, stream>>>(Wq, sel, Wqb);
    cvt_w_t_bf16<<<dim3(nw / 256), 256, 0, stream>>>(Wk, sel, Wkb);
    cvt_w_t_bf16<<<dim3(nw / 256), 256, 0, stream>>>(Wv, sel, Wvb);
    cvt_w_t_bf16<<<dim3(nw / 256), 256, 0, stream>>>(Wo, nullptr, Wob);

    dim3 ggrid(kDIM / 128, kM / 128);         // (4, 128)
    gemm512<<<ggrid, 256, 0, stream>>>(xb, Wqb, nullptr, (void*)Qb, 0);
    gemm512<<<ggrid, 256, 0, stream>>>(xb, Wkb, nullptr, (void*)Kb, 0);
    gemm512<<<ggrid, 256, 0, stream>>>(xb, Wvb, nullptr, (void*)Vb, 2);

    flash64<<<dim3(kSEQ / 64, kBATCH * kHEADS), 128, 0, stream>>>(Qb, Kb, Vb, Ob);

    gemm512<<<ggrid, 256, 0, stream>>>(Ob, Wob, bo, (void*)out, 1);
}
